// MoEFFN_20444044329636
// MI455X (gfx1250) — matrix operand order repaired and checked by reading, not compiled, not run on hardware
//
#include <hip/hip_runtime.h>
#include <stdint.h>

// ---------------- CDNA5 WMMA types ----------------
typedef __attribute__((ext_vector_type(16))) __bf16 v16bf;
typedef __attribute__((ext_vector_type(8)))  float  v8f;

// ---------------- problem dims (fixed by reference) ----------------
#define T_TOK 8192   // B*S tokens
#define H_DIM 1024
#define F_DIM 4096
#define NE    8

// ---------------- GEMM tiling ----------------
#define BM 128
#define BN 64
#define BK 32
#define LDA 40   // BK + 8 pad (u16 elems) -> 80B row stride (16B aligned)
#define LDB 40

// ---------------- gfx1250 feature probes (compile-safe) ----------------
#if defined(__gfx1250__) && __has_builtin(__builtin_amdgcn_global_load_async_to_lds_b128)
#define HAVE_ASYNC_LDS 1
#else
#define HAVE_ASYNC_LDS 0
#endif
#if defined(__gfx1250__) && __has_builtin(__builtin_amdgcn_tensor_load_to_lds) && \
    __has_builtin(__builtin_amdgcn_s_wait_tensorcnt)
#define HAVE_TDM 1
#else
#define HAVE_TDM 0
#endif

union FragU { uint4 u[2]; v16bf v; };

__device__ __forceinline__ unsigned short f2bf(float f) {
  unsigned int u = __float_as_uint(f);
  return (unsigned short)((u + 0x7fffu + ((u >> 16) & 1u)) >> 16);  // RNE
}

// packed fp32x2 -> bf16x2 (v_cvt_pk_bf16_f32 when available)
#if defined(__gfx1250__) && __has_builtin(__builtin_amdgcn_cvt_pk_bf16_f32)
__device__ __forceinline__ unsigned pack_bf16x2(float a, float b) {
  auto r = __builtin_amdgcn_cvt_pk_bf16_f32(a, b);
  union { decltype(r) v; unsigned u; } c;
  c.v = r;
  return c.u;
}
#else
__device__ __forceinline__ unsigned pack_bf16x2(float a, float b) {
  return (unsigned)f2bf(a) | ((unsigned)f2bf(b) << 16);
}
#endif

#if HAVE_ASYNC_LDS
typedef int v4i32_vs __attribute__((vector_size(16)));
typedef __attribute__((address_space(1))) v4i32_vs* gbl_v4i_ptr;
typedef __attribute__((address_space(3))) v4i32_vs* lds_v4i_ptr;
#endif

// 16B global -> LDS copy: async-to-LDS on gfx1250 if available, else sync.
__device__ __forceinline__ void lds_cp_b128(void* lds_dst, const void* gsrc) {
#if HAVE_ASYNC_LDS
  __builtin_amdgcn_global_load_async_to_lds_b128(
      (gbl_v4i_ptr)(uintptr_t)gsrc,
      (lds_v4i_ptr)(uintptr_t)(unsigned)(uintptr_t)lds_dst,
      0, 0);
#else
  *(uint4*)lds_dst = *(const uint4*)gsrc;
#endif
}

__device__ __forceinline__ void moe_async_wait() {
#if HAVE_ASYNC_LDS
#if __has_builtin(__builtin_amdgcn_s_wait_asynccnt)
  __builtin_amdgcn_s_wait_asynccnt(0);
#else
  asm volatile("s_wait_asynccnt 0" ::: "memory");
#endif
#endif
}

#if HAVE_TDM
typedef unsigned int u32x4 __attribute__((ext_vector_type(4)));
typedef int i32x8 __attribute__((ext_vector_type(8)));
typedef int i32x4 __attribute__((ext_vector_type(4)));

// TDM: load a 128x32 bf16 tile (row stride F_DIM elems) into LDS with
// pad_interval=16 dwords (64B row) + pad_amount=4 dwords (16B) -> LDA=40 elems.
__device__ __forceinline__ void tdm_load_tile_128x32(const void* gsrc, unsigned ldsOff) {
  unsigned long long ga = (unsigned long long)(uintptr_t)gsrc;
  u32x4 g0;
  g0[0] = 1u;                                            // count=1, user desc
  g0[1] = ldsOff;                                        // lds_addr (bytes)
  g0[2] = (unsigned)ga;                                  // global_addr[31:0]
  g0[3] = (unsigned)((ga >> 32) & 0x01ffffffu) | (2u << 30);  // addr[56:32] | type=2
  i32x8 g1;
  g1[0] = (int)((1u << 16) | (1u << 20) | (3u << 22) | (3u << 25));
          // data_size=2B | pad_enable | pad_interval=16dw | pad_amount=4dw
  g1[1] = (int)((unsigned)(F_DIM & 0xffffu) << 16);      // tensor_dim0[15:0]
  g1[2] = (int)(((unsigned)F_DIM >> 16) | ((unsigned)(T_TOK & 0xffff) << 16));
  g1[3] = (int)(((unsigned)T_TOK >> 16) | (32u << 16));  // tensor_dim1 hi | tile_dim0=32
  g1[4] = 128;                                           // tile_dim1=128, tile_dim2=0
  g1[5] = F_DIM;                                         // tensor_dim0_stride lo
  g1[6] = 0;
  g1[7] = 0;
  i32x4 z4 = {0, 0, 0, 0};
  i32x8 z8 = {0, 0, 0, 0, 0, 0, 0, 0};
  __builtin_amdgcn_tensor_load_to_lds(g0, g1, z4, z4, z8, 0);
}
#endif

// =====================================================================
// Router: logits -> softmax -> top-2, build compacted per-expert lists
// =====================================================================
__global__ void moe_router_kernel(const float* __restrict__ x,
                                  const float* __restrict__ rw,
                                  int* __restrict__ cnt,
                                  int* __restrict__ idx,
                                  float* __restrict__ gsc) {
  int t = blockIdx.x * blockDim.x + threadIdx.x;
  if (t >= T_TOK) return;
  float logit[NE];
#pragma unroll
  for (int e = 0; e < NE; ++e) logit[e] = 0.f;
  const float* xr = x + (size_t)t * H_DIM;
  for (int h = 0; h < H_DIM; h += 4) {
    float4 xv = *(const float4*)(xr + h);
#pragma unroll
    for (int e = 0; e < NE; ++e) {
      logit[e] += xv.x * rw[(h + 0) * NE + e] + xv.y * rw[(h + 1) * NE + e]
                + xv.z * rw[(h + 2) * NE + e] + xv.w * rw[(h + 3) * NE + e];
    }
  }
  float m = logit[0];
#pragma unroll
  for (int e = 1; e < NE; ++e) m = fmaxf(m, logit[e]);
  float p[NE], s = 0.f;
#pragma unroll
  for (int e = 0; e < NE; ++e) { p[e] = __expf(logit[e] - m); s += p[e]; }
  float inv = 1.f / s;
  int i1 = 0;
#pragma unroll
  for (int e = 1; e < NE; ++e) if (p[e] > p[i1]) i1 = e;
  int i2 = (i1 == 0) ? 1 : 0;
#pragma unroll
  for (int e = 0; e < NE; ++e) if (e != i1 && p[e] > p[i2]) i2 = e;

  int pos1 = atomicAdd(&cnt[i1], 1);
  idx[i1 * T_TOK + pos1] = t;
  gsc[i1 * T_TOK + pos1] = p[i1] * inv;
  int pos2 = atomicAdd(&cnt[i2], 1);
  idx[i2 * T_TOK + pos2] = t;
  gsc[i2 * T_TOK + pos2] = p[i2] * inv;
}

// =====================================================================
// x (fp32) -> bf16 activation buffer
// =====================================================================
__global__ void moe_cvt_bf16_kernel(const float* __restrict__ x,
                                    unsigned short* __restrict__ xb, int n) {
  int i = (blockIdx.x * blockDim.x + threadIdx.x) * 4;
  if (i >= n) return;
  float4 v = *(const float4*)(x + i);
  uint2 r;
  r.x = pack_bf16x2(v.x, v.y);
  r.y = pack_bf16x2(v.z, v.w);
  *(uint2*)(xb + i) = r;
}

// B-tile helper: fp32 float4 -> bf16 (packed cvt), transposed [n][k] store
__device__ __forceinline__ void storeB(unsigned short (*Bs)[LDB], int n4, int kk, float4 v) {
  unsigned p01 = pack_bf16x2(v.x, v.y);
  unsigned p23 = pack_bf16x2(v.z, v.w);
  Bs[n4 + 0][kk] = (unsigned short)p01;
  Bs[n4 + 1][kk] = (unsigned short)(p01 >> 16);
  Bs[n4 + 2][kk] = (unsigned short)p23;
  Bs[n4 + 3][kk] = (unsigned short)(p23 >> 16);
}

// =====================================================================
// GEMM1+2 fused: h = silu(x@w1 + b1) * (x@w2 + b2) for expert e
// double-buffered LDS, async-to-LDS A gather, register-staged B
// =====================================================================
__global__ __launch_bounds__(256)
void moe_gemm12_swiglu_kernel(const unsigned short* __restrict__ xb,
                              const float* __restrict__ w1,
                              const float* __restrict__ b1,
                              const float* __restrict__ w2,
                              const float* __restrict__ b2,
                              const int* __restrict__ cnt,
                              const int* __restrict__ idx,
                              unsigned short* __restrict__ hb,
                              int e) {
  const int Me = cnt[e];
  const int rowBase = blockIdx.y * BM;
  if (rowBase >= Me) return;
  const int n0 = blockIdx.x * BN;

  __shared__ unsigned short As[2][BM][LDA];
  __shared__ unsigned short Bs1[2][BN][LDB];
  __shared__ unsigned short Bs2[2][BN][LDB];
  __shared__ int tokS[BM];

  const int tid  = threadIdx.x;
  const int lane = tid & 31, wid = tid >> 5;
  const int waveM = wid & 3, waveN = wid >> 2;
  const int hsel = lane >> 4, lm = lane & 15;

  if (tid < BM) {
    int r = rowBase + tid;
    tokS[tid] = idx[e * T_TOK + ((r < Me) ? r : (Me - 1))];
  }
  __syncthreads();

  const float* w1e = w1 + (size_t)e * H_DIM * F_DIM;
  const float* w2e = w2 + (size_t)e * H_DIM * F_DIM;

  // per-thread A gather addresses, hoisted out of the K loop
  const int aRow0 = tid >> 2, aRow1 = aRow0 + 64;
  const int aC8 = (tid & 3) * 8;
  const unsigned short* aSrc0 = xb + (size_t)tokS[aRow0] * H_DIM + aC8;
  const unsigned short* aSrc1 = xb + (size_t)tokS[aRow1] * H_DIM + aC8;

  // per-thread B addresses
  const int bK0 = tid >> 4, bK1 = bK0 + 16;
  const int bN4 = (tid & 15) * 4;
  const float* b1p0 = w1e + (size_t)bK0 * F_DIM + n0 + bN4;
  const float* b1p1 = w1e + (size_t)bK1 * F_DIM + n0 + bN4;
  const float* b2p0 = w2e + (size_t)bK0 * F_DIM + n0 + bN4;
  const float* b2p1 = w2e + (size_t)bK1 * F_DIM + n0 + bN4;

  // ---- prologue: stage tile 0 ----
  lds_cp_b128(&As[0][aRow0][aC8], aSrc0);
  lds_cp_b128(&As[0][aRow1][aC8], aSrc1);
  {
    float4 r10 = *(const float4*)b1p0, r11 = *(const float4*)b1p1;
    float4 r20 = *(const float4*)b2p0, r21 = *(const float4*)b2p1;
    storeB(Bs1[0], bN4, bK0, r10); storeB(Bs1[0], bN4, bK1, r11);
    storeB(Bs2[0], bN4, bK0, r20); storeB(Bs2[0], bN4, bK1, r21);
  }
  moe_async_wait();
  __syncthreads();

  v8f accG[2][2], accU[2][2];
#pragma unroll
  for (int i = 0; i < 2; ++i)
#pragma unroll
    for (int j = 0; j < 2; ++j)
#pragma unroll
      for (int r = 0; r < 8; ++r) { accG[i][j][r] = 0.f; accU[i][j][r] = 0.f; }

  int buf = 0;
  for (int k0 = 0; k0 < H_DIM; k0 += BK) {
    const int nxt = buf ^ 1;
    const bool hasNext = (k0 + BK) < H_DIM;

    // 1) kick off next A tile (async -> LDS), overlaps everything below
    if (hasNext) {
      lds_cp_b128(&As[nxt][aRow0][aC8], aSrc0 + (k0 + BK));
      lds_cp_b128(&As[nxt][aRow1][aC8], aSrc1 + (k0 + BK));
    }

    // 2) fragment loads from current buffers (ISA VGPR layouts, b128 pairs)
    FragU a[2], fb1[2], fb2[2];
#pragma unroll
    for (int mi = 0; mi < 2; ++mi) {
      int m = waveM * 32 + mi * 16 + lm;
      a[mi].u[0] = *(const uint4*)&As[buf][m][hsel * 8];
      a[mi].u[1] = *(const uint4*)&As[buf][m][16 + hsel * 8];
    }
#pragma unroll
    for (int ni = 0; ni < 2; ++ni) {
      int n = waveN * 32 + ni * 16 + lm;
      fb1[ni].u[0] = *(const uint4*)&Bs1[buf][n][hsel * 8];
      fb1[ni].u[1] = *(const uint4*)&Bs1[buf][n][16 + hsel * 8];
      fb2[ni].u[0] = *(const uint4*)&Bs2[buf][n][hsel * 8];
      fb2[ni].u[1] = *(const uint4*)&Bs2[buf][n][16 + hsel * 8];
    }

    // 3) issue next B tile global loads into registers (waited after WMMA)
    float4 r10{}, r11{}, r20{}, r21{};
    if (hasNext) {
      size_t ko = (size_t)(k0 + BK) * F_DIM;
      r10 = *(const float4*)(b1p0 + ko); r11 = *(const float4*)(b1p1 + ko);
      r20 = *(const float4*)(b2p0 + ko); r21 = *(const float4*)(b2p1 + ko);
      __builtin_prefetch(b1p0 + ko + (size_t)BK * F_DIM, 0, 1);
      __builtin_prefetch(b2p0 + ko + (size_t)BK * F_DIM, 0, 1);
    }

    // 4) matrix math on current tile
#pragma unroll
    for (int mi = 0; mi < 2; ++mi)
#pragma unroll
      for (int ni = 0; ni < 2; ++ni) {
        accG[mi][ni] = __builtin_amdgcn_wmma_f32_16x16x32_bf16(
            false, a[mi].v, false, fb1[ni].v, (short)0, accG[mi][ni], false, false);
        accU[mi][ni] = __builtin_amdgcn_wmma_f32_16x16x32_bf16(
            false, a[mi].v, false, fb2[ni].v, (short)0, accU[mi][ni], false, false);
      }

    // 5) convert+store next B tile into the other buffer
    if (hasNext) {
      storeB(Bs1[nxt], bN4, bK0, r10); storeB(Bs1[nxt], bN4, bK1, r11);
      storeB(Bs2[nxt], bN4, bK0, r20); storeB(Bs2[nxt], bN4, bK1, r21);
    }

    moe_async_wait();
    __syncthreads();
    buf = nxt;
  }

  // ---- epilogue: SwiGLU, write bf16 h (compacted rows) ----
  const float* b1e = b1 + e * F_DIM;
  const float* b2e = b2 + e * F_DIM;
#pragma unroll
  for (int mi = 0; mi < 2; ++mi)
#pragma unroll
    for (int ni = 0; ni < 2; ++ni) {
      int col = n0 + waveN * 32 + ni * 16 + lm;
      float bg = b1e[col], bu = b2e[col];
#pragma unroll
      for (int r = 0; r < 8; r += 2) {
        int m0 = waveM * 32 + mi * 16 + hsel * 8 + r;
        int gr0 = rowBase + m0, gr1 = gr0 + 1;
        float g0 = accG[mi][ni][r] + bg,     u0 = accU[mi][ni][r] + bu;
        float g1 = accG[mi][ni][r + 1] + bg, u1 = accU[mi][ni][r + 1] + bu;
        float hv0 = (g0 / (1.f + __expf(-g0))) * u0;  // silu(g)*u
        float hv1 = (g1 / (1.f + __expf(-g1))) * u1;
        unsigned p = pack_bf16x2(hv0, hv1);
        if (gr0 < Me) hb[(size_t)gr0 * F_DIM + col] = (unsigned short)p;
        if (gr1 < Me) hb[(size_t)gr1 * F_DIM + col] = (unsigned short)(p >> 16);
      }
    }
}

// =====================================================================
// GEMM3: y = h @ w3 + b3; scatter out[token] += prob * y  (expert e)
// A tile via Tensor Data Mover (contiguous 128x32 bf16), double-buffered
// =====================================================================
__global__ __launch_bounds__(256)
void moe_gemm3_scatter_kernel(const unsigned short* __restrict__ hb,
                              const float* __restrict__ w3,
                              const float* __restrict__ b3,
                              const int* __restrict__ cnt,
                              const int* __restrict__ idx,
                              const float* __restrict__ gsc,
                              float* __restrict__ out,
                              int e) {
  const int Me = cnt[e];
  const int rowBase = blockIdx.y * BM;
  if (rowBase >= Me) return;
  const int n0 = blockIdx.x * BN;  // in H

  __shared__ unsigned short As[2][BM][LDA];
  __shared__ unsigned short Bs[2][BN][LDB];
  __shared__ int   tokS[BM];
  __shared__ float pS[BM];

  const int tid  = threadIdx.x;
  const int lane = tid & 31, wid = tid >> 5;
  const int waveM = wid & 3, waveN = wid >> 2;
  const int hsel = lane >> 4, lm = lane & 15;

  if (tid < BM) {
    int r = rowBase + tid;
    int rc = (r < Me) ? r : (Me - 1);
    tokS[tid] = idx[e * T_TOK + rc];
    pS[tid]   = gsc[e * T_TOK + rc];
  }
  __syncthreads();

  const float* w3e = w3 + (size_t)e * F_DIM * H_DIM;
  const unsigned short* aBase = hb + (size_t)rowBase * F_DIM;

#if !HAVE_TDM
  const int aRow0 = tid >> 2, aRow1 = aRow0 + 64;
  const int aC8 = (tid & 3) * 8;
#endif
  const int bK0 = tid >> 4, bK1 = bK0 + 16;
  const int bN4 = (tid & 15) * 4;
  const float* b3p0 = w3e + (size_t)bK0 * H_DIM + n0 + bN4;
  const float* b3p1 = w3e + (size_t)bK1 * H_DIM + n0 + bN4;

  // ---- prologue: stage tile 0 ----
#if HAVE_TDM
  if (wid == 0)
    tdm_load_tile_128x32(aBase, (unsigned)(uintptr_t)&As[0][0][0]);
#else
  lds_cp_b128(&As[0][aRow0][aC8], aBase + (size_t)aRow0 * F_DIM + aC8);
  lds_cp_b128(&As[0][aRow1][aC8], aBase + (size_t)aRow1 * F_DIM + aC8);
#endif
  {
    float4 r0 = *(const float4*)b3p0, r1 = *(const float4*)b3p1;
    storeB(Bs[0], bN4, bK0, r0); storeB(Bs[0], bN4, bK1, r1);
  }
#if HAVE_TDM
  __builtin_amdgcn_s_wait_tensorcnt(0);
#else
  moe_async_wait();
#endif
  __syncthreads();

  v8f acc[2][2];
#pragma unroll
  for (int i = 0; i < 2; ++i)
#pragma unroll
    for (int j = 0; j < 2; ++j)
#pragma unroll
      for (int r = 0; r < 8; ++r) acc[i][j][r] = 0.f;

  int buf = 0;
  for (int k0 = 0; k0 < F_DIM; k0 += BK) {
    const int nxt = buf ^ 1;
    const bool hasNext = (k0 + BK) < F_DIM;

    // 1) kick off next A tile
    if (hasNext) {
#if HAVE_TDM
      if (wid == 0)
        tdm_load_tile_128x32(aBase + (k0 + BK), (unsigned)(uintptr_t)&As[nxt][0][0]);
#else
      lds_cp_b128(&As[nxt][aRow0][aC8], aBase + (size_t)aRow0 * F_DIM + k0 + BK + aC8);
      lds_cp_b128(&As[nxt][aRow1][aC8], aBase + (size_t)aRow1 * F_DIM + k0 + BK + aC8);
#endif
    }

    // 2) fragment loads
    FragU a[2], fb[2];
#pragma unroll
    for (int mi = 0; mi < 2; ++mi) {
      int m = waveM * 32 + mi * 16 + lm;
      a[mi].u[0] = *(const uint4*)&As[buf][m][hsel * 8];
      a[mi].u[1] = *(const uint4*)&As[buf][m][16 + hsel * 8];
    }
#pragma unroll
    for (int ni = 0; ni < 2; ++ni) {
      int n = waveN * 32 + ni * 16 + lm;
      fb[ni].u[0] = *(const uint4*)&Bs[buf][n][hsel * 8];
      fb[ni].u[1] = *(const uint4*)&Bs[buf][n][16 + hsel * 8];
    }

    // 3) next B tile into registers
    float4 r0{}, r1{};
    if (hasNext) {
      size_t ko = (size_t)(k0 + BK) * H_DIM;
      r0 = *(const float4*)(b3p0 + ko);
      r1 = *(const float4*)(b3p1 + ko);
      __builtin_prefetch(b3p0 + ko + (size_t)BK * H_DIM, 0, 1);
    }

    // 4) matrix math
#pragma unroll
    for (int mi = 0; mi < 2; ++mi)
#pragma unroll
      for (int ni = 0; ni < 2; ++ni)
        acc[mi][ni] = __builtin_amdgcn_wmma_f32_16x16x32_bf16(
            false, a[mi].v, false, fb[ni].v, (short)0, acc[mi][ni], false, false);

    // 5) store next B tile
    if (hasNext) { storeB(Bs[nxt], bN4, bK0, r0); storeB(Bs[nxt], bN4, bK1, r1); }

#if HAVE_TDM
    __builtin_amdgcn_s_wait_tensorcnt(0);
#else
    moe_async_wait();
#endif
    __syncthreads();
    buf = nxt;
  }

  // ---- epilogue: prob-scaled scatter add (experts serialized on stream) ----
  const float* b3e = b3 + e * H_DIM;
#pragma unroll
  for (int mi = 0; mi < 2; ++mi)
#pragma unroll
    for (int ni = 0; ni < 2; ++ni) {
      int col = n0 + waveN * 32 + ni * 16 + lm;
      float bb = b3e[col];
#pragma unroll
      for (int r = 0; r < 8; ++r) {
        int m  = waveM * 32 + mi * 16 + hsel * 8 + r;
        int gr = rowBase + m;
        if (gr < Me) {
          float y = acc[mi][ni][r] + bb;
          float* o = out + (size_t)tokS[m] * H_DIM + col;
          *o += pS[m] * y;
        }
      }
    }
}

// =====================================================================
// host-side launcher
// =====================================================================
extern "C" void kernel_launch(void* const* d_in, const int* in_sizes, int n_in,
                              void* d_out, int out_size, void* d_ws, size_t ws_size,
                              hipStream_t stream) {
  (void)in_sizes; (void)n_in; (void)ws_size;
  const float* x  = (const float*)d_in[0];
  const float* rw = (const float*)d_in[1];
  const float* w1 = (const float*)d_in[2];
  const float* b1 = (const float*)d_in[3];
  const float* w2 = (const float*)d_in[4];
  const float* b2 = (const float*)d_in[5];
  const float* w3 = (const float*)d_in[6];
  const float* b3 = (const float*)d_in[7];
  float* out = (float*)d_out;

  char* ws = (char*)d_ws;
  size_t off = 0;
  int*   cnt = (int*)(ws + off);            off += 256;
  int*   idx = (int*)(ws + off);            off += (size_t)NE * T_TOK * sizeof(int);
  float* gsc = (float*)(ws + off);          off += (size_t)NE * T_TOK * sizeof(float);
  unsigned short* xb = (unsigned short*)(ws + off); off += (size_t)T_TOK * H_DIM * 2;
  unsigned short* hb = (unsigned short*)(ws + off); off += (size_t)T_TOK * F_DIM * 2;

  (void)hipMemsetAsync(cnt, 0, NE * sizeof(int), stream);
  (void)hipMemsetAsync(d_out, 0, (size_t)out_size * sizeof(float), stream);

  moe_router_kernel<<<T_TOK / 256, 256, 0, stream>>>(x, rw, cnt, idx, gsc);
  moe_cvt_bf16_kernel<<<(T_TOK * H_DIM / 4) / 256, 256, 0, stream>>>(x, xb, T_TOK * H_DIM);

  dim3 g12(F_DIM / BN, T_TOK / BM);  // 64 x 64
  dim3 g3 (H_DIM / BN, T_TOK / BM);  // 16 x 64
  for (int e = 0; e < NE; ++e) {
    moe_gemm12_swiglu_kernel<<<g12, 256, 0, stream>>>(xb, w1, b1, w2, b2, cnt, idx, hb, e);
    moe_gemm3_scatter_kernel<<<g3, 256, 0, stream>>>(hb, w3, b3, cnt, idx, gsc, out, e);
  }
}
